// GATv4_47141561041256
// MI455X (gfx1250) — compile-verified, matrix operand order checked by reading, not run on hardware
//
#include <hip/hip_runtime.h>
#include <hip/hip_bf16.h>

// ---------------------------------------------------------------------------
// GATv4 forward for MI455X (gfx1250, wave32).
//  - GEMMs on v_wmma_f32_16x16x32_bf16, tiles staged by the Tensor Data Mover
//    (tensor_load_to_lds + s_wait_tensorcnt, double buffered) when available.
//  - Activations/weights pre-converted to bf16 so TDM tiles are raw copies.
//  - Edge softmax/aggregation is gather-based over a CSR built per call.
// ---------------------------------------------------------------------------

#define NN     16384        // nodes
#define DIN    1024
#define HCH    512          // H*C
#define NHEAD  4
#define NC     128          // channels per head
#define NE     262144       // edges (before self loops)
#define NEP    (NE + NN)    // edges incl. self loops
#define NBATCH 16
#define NODES  1024         // nodes per graph

typedef __bf16 bf16_t;
typedef __attribute__((ext_vector_type(16))) __bf16 v16bf;
typedef __attribute__((ext_vector_type(8)))  __bf16 v8bf;
typedef __attribute__((ext_vector_type(8)))  float  v8f;
typedef __attribute__((ext_vector_type(4)))  unsigned int u32x4;
typedef __attribute__((ext_vector_type(8)))  int i32x8;
typedef __attribute__((ext_vector_type(4)))  int i32x4;

struct alignas(8)  BF4 { bf16_t v[4]; };
struct alignas(16) BF8 { bf16_t v[8]; };

__device__ __forceinline__ float eluf(float v)  { return v > 0.f ? v : (expf(v) - 1.f); }
__device__ __forceinline__ float lrelu(float v) { return v > 0.f ? v : 0.2f * v; }

#if __has_builtin(__builtin_amdgcn_tensor_load_to_lds)
#define HAVE_TDM 1
#else
#define HAVE_TDM 0
#endif

// ---------------------------------------------------------------------------
// TDM 2D tile load: tileH rows of tileW bf16 elements (row stride = rowElems),
// into LDS with +16B padding after each 64B row (PITCH=40 bf16).
// D# layout per CDNA5 ISA §8.3/8.4 (group0 128b, group1 256b).
// ---------------------------------------------------------------------------
#if HAVE_TDM
__device__ __forceinline__ void tdm_tile_load(unsigned lds_addr, const void* gptr,
                                              unsigned rowElems, unsigned tileW,
                                              unsigned tileH)
{
    unsigned long long ga = (unsigned long long)(size_t)gptr;
    u32x4 g0;
    g0[0] = 1u;                                   // count=1 (valid), user mode
    g0[1] = lds_addr;                             // lds_addr (bytes)
    g0[2] = (unsigned)(ga & 0xffffffffu);         // global_addr[31:0]
    g0[3] = (unsigned)((ga >> 32) & 0x01ffffffu)  // global_addr[56:32]
          | (2u << 30);                           // type = 2 ("image")
    i32x8 g1;
    g1[0] = (int)((1u << 16)                      // data_size = 2 bytes
                | (1u << 20)                      // pad_enable
                | (3u << 22)                      // pad_interval: 16 DWORDs
                | (3u << 25));                    // pad_amount:   4 DWORDs
    g1[1] = (int)((tileW & 0xffffu) << 16);                         // tensor_dim0 lo16
    g1[2] = (int)((tileW >> 16) | ((tileH & 0xffffu) << 16));       // dim0 hi | dim1 lo
    g1[3] = (int)((tileH >> 16) | ((tileW & 0xffffu) << 16));       // dim1 hi | tile_dim0
    g1[4] = (int)(tileH & 0xffffu);                                 // tile_dim1 (dim2=0)
    g1[5] = (int)rowElems;                                          // dim0_stride lo32
    g1[6] = 0;
    g1[7] = 0;
    i32x4 z4 = {0, 0, 0, 0};
#if __clang_major__ >= 23
    i32x8 z8 = {0, 0, 0, 0, 0, 0, 0, 0};
    __builtin_amdgcn_tensor_load_to_lds(g0, g1, z4, z4, z8, 0);
#else
    __builtin_amdgcn_tensor_load_to_lds(g0, g1, z4, z4, 0);
#endif
}
#endif

// ---------------------------------------------------------------------------
// WMMA GEMM: C[M,NCOL] = A[M,K] @ Bt[NCOL,K]^T, bf16 in, f32 out.
// Block tile 128x128, BK=32, 256 threads = 8 waves, wave tile 32x64 (2x4 wmma).
// NCOL is a template constant so the epilogue folds row strides into
// immediate store offsets (no v_mul_u64 address chains).
// ---------------------------------------------------------------------------
#define BM 128
#define BN 128
#define BK 32
#define PITCH 40   // bf16 elements per LDS row (80B: conflict-free b128 reads)

template <int NCOL>
__global__ __launch_bounds__(256)
void wmma_gemm_bf16(const bf16_t* __restrict__ A, const bf16_t* __restrict__ Bt,
                    float* __restrict__ C, int K)
{
#if HAVE_TDM
    __shared__ __align__(16) bf16_t sA[2][BM * PITCH];
    __shared__ __align__(16) bf16_t sB[2][BN * PITCH];
#else
    __shared__ __align__(16) bf16_t sA[1][BM * PITCH];
    __shared__ __align__(16) bf16_t sB[1][BN * PITCH];
#endif
    const int t     = threadIdx.x;
    const int lane  = t & 31;
    const int wid   = t >> 5;
    const int waveM = wid & 3;     // 4 waves x 32 rows
    const int waveN = wid >> 2;    // 2 waves x 64 cols
    const int m0    = blockIdx.y * BM;
    const int n0    = blockIdx.x * BN;
    const int half  = lane >> 4;
    const int rsel  = lane & 15;

    v8f acc[2][4] = {};
    const int steps = K / BK;

#if HAVE_TDM
    if (wid == 0) {
        tdm_tile_load((unsigned)(size_t)&sA[0][0], A + (size_t)m0 * K, K, BK, BM);
        tdm_tile_load((unsigned)(size_t)&sB[0][0], Bt + (size_t)n0 * K, K, BK, BN);
        __builtin_amdgcn_s_wait_tensorcnt(0);
    }
    __syncthreads();
#endif

    for (int s = 0; s < steps; ++s) {
        const int k0 = s * BK;
#if HAVE_TDM
        const int cur = s & 1;
        if (wid == 0 && s + 1 < steps) {   // async: next tile into other buffer
            tdm_tile_load((unsigned)(size_t)&sA[cur ^ 1][0],
                          A + (size_t)m0 * K + (k0 + BK), K, BK, BM);
            tdm_tile_load((unsigned)(size_t)&sB[cur ^ 1][0],
                          Bt + (size_t)n0 * K + (k0 + BK), K, BK, BN);
        }
#else
        const int cur = 0;
        uint4 ra[2], rb[2];                // batched loads, then stores
        #pragma unroll
        for (int p = 0; p < 2; ++p) {
            int cc = t + p * 256, row = cc >> 2, q = cc & 3;
            ra[p] = *(const uint4*)(A + (size_t)(m0 + row) * K + k0 + q * 8);
            rb[p] = *(const uint4*)(Bt + (size_t)(n0 + row) * K + k0 + q * 8);
        }
        __syncthreads();
        #pragma unroll
        for (int p = 0; p < 2; ++p) {
            int cc = t + p * 256, row = cc >> 2, q = cc & 3;
            *(uint4*)(&sA[0][row * PITCH + q * 8]) = ra[p];
            *(uint4*)(&sB[0][row * PITCH + q * 8]) = rb[p];
        }
        __syncthreads();
#endif
        const bf16_t* pa = &sA[cur][0];
        const bf16_t* pb = &sB[cur][0];

        // fragments per ISA 16-bit 16x32 layout:
        // lanes 0-15: K runs [0..7],[16..23]; lanes 16-31: [8..15],[24..31]
        v16bf af[2], bq[4];
        #pragma unroll
        for (int i = 0; i < 2; ++i) {
            const bf16_t* base = pa + (waveM * 32 + i * 16 + rsel) * PITCH;
            v8bf lo = *(const v8bf*)(base + half * 8);
            v8bf hi = *(const v8bf*)(base + 16 + half * 8);
            af[i] = __builtin_shufflevector(lo, hi, 0,1,2,3,4,5,6,7,8,9,10,11,12,13,14,15);
        }
        #pragma unroll
        for (int j = 0; j < 4; ++j) {
            const bf16_t* base = pb + (waveN * 64 + j * 16 + rsel) * PITCH;
            v8bf lo = *(const v8bf*)(base + half * 8);
            v8bf hi = *(const v8bf*)(base + 16 + half * 8);
            bq[j] = __builtin_shufflevector(lo, hi, 0,1,2,3,4,5,6,7,8,9,10,11,12,13,14,15);
        }
        #pragma unroll
        for (int i = 0; i < 2; ++i)
            #pragma unroll
            for (int j = 0; j < 4; ++j)
                acc[i][j] = __builtin_amdgcn_wmma_f32_16x16x32_bf16(
                    false, af[i], false, bq[j], (short)0, acc[i][j], false, false);

#if HAVE_TDM
        if (wid == 0 && s + 1 < steps) __builtin_amdgcn_s_wait_tensorcnt(0);
        __syncthreads();
#endif
    }

    // store C per f32 16x16 layout: VGPR r -> M = r + 8*(lane>=16).
    // NCOL is constexpr: r*NCOL folds into the store's immediate offset.
    #pragma unroll
    for (int i = 0; i < 2; ++i)
        #pragma unroll
        for (int j = 0; j < 4; ++j) {
            int row0 = m0 + waveM * 32 + i * 16 + ((lane >> 4) << 3);
            int col  = n0 + waveN * 64 + j * 16 + (lane & 15);
            float* cp = C + (size_t)row0 * NCOL + col;
            #pragma unroll
            for (int r = 0; r < 8; ++r)
                cp[(size_t)r * NCOL] = acc[i][j][r];
        }
}

// ---------------------------------------------------------------------------
// f32 -> bf16 bulk convert (8 elems/thread, 16B stores)
// ---------------------------------------------------------------------------
__global__ __launch_bounds__(256)
void k_f32_to_bf16(const float* __restrict__ in, bf16_t* __restrict__ out, int n8)
{
    int i = blockIdx.x * blockDim.x + threadIdx.x;
    if (i >= n8) return;
    const float4 a = *(const float4*)(in + (size_t)i * 8);
    const float4 b = *(const float4*)(in + (size_t)i * 8 + 4);
    BF8 r;
    r.v[0] = (bf16_t)a.x; r.v[1] = (bf16_t)a.y; r.v[2] = (bf16_t)a.z; r.v[3] = (bf16_t)a.w;
    r.v[4] = (bf16_t)b.x; r.v[5] = (bf16_t)b.y; r.v[6] = (bf16_t)b.z; r.v[7] = (bf16_t)b.w;
    *(BF8*)(out + (size_t)i * 8) = r;
}

// ---------------------------------------------------------------------------
// Wt[n][k] = (bf16) W[k][n]  (32x32 LDS-tiled transpose)
// ---------------------------------------------------------------------------
__global__ __launch_bounds__(256)
void k_transpose_bf16(const float* __restrict__ W, bf16_t* __restrict__ Wt,
                      int K, int Ncol)
{
    __shared__ float tile[32][33];
    int bx = blockIdx.x * 32;   // n base
    int by = blockIdx.y * 32;   // k base
    int tx = threadIdx.x, ty = threadIdx.y;
    for (int i = ty; i < 32; i += 8)
        tile[i][tx] = W[(size_t)(by + i) * Ncol + bx + tx];
    __syncthreads();
    for (int i = ty; i < 32; i += 8)
        Wt[(size_t)(bx + i) * K + by + tx] = (bf16_t)tile[tx][i];
}

// ---------------------------------------------------------------------------
// CSR build (destination-major), self loops included via counts init = 1
// ---------------------------------------------------------------------------
__global__ void k_init_counts(int* __restrict__ counts)
{
    int n = blockIdx.x * blockDim.x + threadIdx.x;
    if (n < NN) counts[n] = 1;
}

__global__ void k_count_edges(const int* __restrict__ ei, int* __restrict__ counts)
{
    int e = blockIdx.x * blockDim.x + threadIdx.x;
    if (e < NE) atomicAdd(&counts[ei[NE + e]], 1);
}

__global__ __launch_bounds__(1024)
void k_scan16k(const int* __restrict__ counts, int* __restrict__ rowptr,
               int* __restrict__ cursor)
{
    __shared__ int part[1024];
    int t = threadIdx.x;
    int base = t * 16;
    int loc[16];
    int s = 0;
    #pragma unroll
    for (int i = 0; i < 16; ++i) { loc[i] = counts[base + i]; s += loc[i]; }
    part[t] = s;
    __syncthreads();
    for (int off = 1; off < 1024; off <<= 1) {
        int v = (t >= off) ? part[t - off] : 0;
        __syncthreads();
        part[t] += v;
        __syncthreads();
    }
    int run = (t == 0) ? 0 : part[t - 1];
    #pragma unroll
    for (int i = 0; i < 16; ++i) {
        rowptr[base + i] = run;
        cursor[base + i] = run;
        run += loc[i];
    }
    if (t == 1023) rowptr[NN] = run;
}

__global__ void k_scatter_self(int* __restrict__ cursor, int* __restrict__ colidx)
{
    int n = blockIdx.x * blockDim.x + threadIdx.x;
    if (n < NN) { int pos = atomicAdd(&cursor[n], 1); colidx[pos] = n; }
}

__global__ void k_scatter_edges(const int* __restrict__ ei, int* __restrict__ cursor,
                                int* __restrict__ colidx)
{
    int e = blockIdx.x * blockDim.x + threadIdx.x;
    if (e < NE) {
        int d = ei[NE + e];
        int pos = atomicAdd(&cursor[d], 1);
        colidx[pos] = ei[e];
    }
}

// ---------------------------------------------------------------------------
// alpha_src / alpha_dst: one wave per node; lane covers 16 contiguous channels
// ---------------------------------------------------------------------------
__global__ __launch_bounds__(256)
void k_gat_alpha(const float* __restrict__ h, const float* __restrict__ attS,
                 const float* __restrict__ attD, float* __restrict__ asr,
                 float* __restrict__ adt)
{
    int wave = (blockIdx.x * blockDim.x + threadIdx.x) >> 5;
    int lane = threadIdx.x & 31;
    if (wave >= NN) return;
    int head  = lane >> 3;
    int cbase = head * NC + (lane & 7) * 16;
    const float* hp = h + (size_t)wave * HCH + lane * 16;
    float ss = 0.f, sd = 0.f;
    #pragma unroll
    for (int i = 0; i < 16; i += 4) {
        float4 hv = *(const float4*)(hp + i);
        float4 sv = *(const float4*)(attS + cbase + i);
        float4 dv = *(const float4*)(attD + cbase + i);
        ss += hv.x * sv.x + hv.y * sv.y + hv.z * sv.z + hv.w * sv.w;
        sd += hv.x * dv.x + hv.y * dv.y + hv.z * dv.z + hv.w * dv.w;
    }
    for (int o = 1; o < 8; o <<= 1) { ss += __shfl_xor(ss, o); sd += __shfl_xor(sd, o); }
    if ((lane & 7) == 0) {
        asr[wave * NHEAD + head] = ss;
        adt[wave * NHEAD + head] = sd;
    }
}

// ---------------------------------------------------------------------------
// Gather-based segment softmax + aggregation + bias + ELU.
// One wave per (node, head). Optionally emits a bf16 copy (next GEMM's A).
// ---------------------------------------------------------------------------
__global__ __launch_bounds__(256)
void k_gat_aggregate(const float* __restrict__ h, const float* __restrict__ asr,
                     const float* __restrict__ adt, const int* __restrict__ rowptr,
                     const int* __restrict__ colidx, const float* __restrict__ bias,
                     float* __restrict__ outp, bf16_t* __restrict__ outb)
{
    int wave = (blockIdx.x * blockDim.x + threadIdx.x) >> 5;
    int lane = threadIdx.x & 31;
    int n  = wave >> 2;
    int hd = wave & 3;
    if (n >= NN) return;
    int start = rowptr[n], end = rowptr[n + 1];
    float ad = adt[n * NHEAD + hd];

    float mx = -3.4e38f;
    for (int j = start + lane; j < end; j += 32)
        mx = fmaxf(mx, lrelu(asr[colidx[j] * NHEAD + hd] + ad));
    for (int o = 16; o > 0; o >>= 1) mx = fmaxf(mx, __shfl_xor(mx, o));

    float den = 0.f;
    for (int j = start + lane; j < end; j += 32)
        den += expf(lrelu(asr[colidx[j] * NHEAD + hd] + ad) - mx);
    for (int o = 16; o > 0; o >>= 1) den += __shfl_xor(den, o);
    float inv = 1.f / (den + 1e-16f);

    float4 acc = make_float4(0.f, 0.f, 0.f, 0.f);
    for (int j = start; j < end; ++j) {        // uniform loop: lanes split channels
        int src = colidx[j];
        float w = expf(lrelu(asr[src * NHEAD + hd] + ad) - mx) * inv;
        const float4 hv = *(const float4*)(h + (size_t)src * HCH + hd * NC + lane * 4);
        acc.x += w * hv.x; acc.y += w * hv.y; acc.z += w * hv.z; acc.w += w * hv.w;
    }
    const float4 bv = *(const float4*)(bias + hd * NC + lane * 4);
    float4 r;
    r.x = eluf(acc.x + bv.x); r.y = eluf(acc.y + bv.y);
    r.z = eluf(acc.z + bv.z); r.w = eluf(acc.w + bv.w);
    size_t o4 = (size_t)n * HCH + hd * NC + lane * 4;
    *(float4*)(outp + o4) = r;
    if (outb) {
        BF4 rb_;
        rb_.v[0] = (bf16_t)r.x; rb_.v[1] = (bf16_t)r.y;
        rb_.v[2] = (bf16_t)r.z; rb_.v[3] = (bf16_t)r.w;
        *(BF4*)(outb + o4) = rb_;
    }
}

// ---------------------------------------------------------------------------
// pooling: x2[n] = dot(h2[n,:512], p2w) + p2b   (one wave per node)
// ---------------------------------------------------------------------------
__global__ __launch_bounds__(256)
void k_pool(const float* __restrict__ h2, const float* __restrict__ pw,
            const float* __restrict__ pb, float* __restrict__ x2)
{
    int wave = (blockIdx.x * blockDim.x + threadIdx.x) >> 5;
    int lane = threadIdx.x & 31;
    if (wave >= NN) return;
    const float* hp = h2 + (size_t)wave * HCH + lane * 16;
    const float* wp = pw + lane * 16;
    float s = 0.f;
    #pragma unroll
    for (int i = 0; i < 16; i += 4) {
        float4 hv = *(const float4*)(hp + i);
        float4 wv = *(const float4*)(wp + i);
        s += hv.x * wv.x + hv.y * wv.y + hv.z * wv.z + hv.w * wv.w;
    }
    for (int o = 16; o > 0; o >>= 1) s += __shfl_xor(s, o);
    if (lane == 0) x2[wave] = s + pb[0];
}

// ---------------------------------------------------------------------------
// layer norm over 16 rows of 1024 (gamma/beta indexed by node position)
// ---------------------------------------------------------------------------
__global__ __launch_bounds__(256)
void k_layernorm(const float* __restrict__ x2, const float* __restrict__ g,
                 const float* __restrict__ b, float* __restrict__ out)
{
    __shared__ float red[256];
    __shared__ float red2[256];
    int row = blockIdx.x, t = threadIdx.x;
    float s = 0.f, s2 = 0.f;
    for (int i = t; i < NODES; i += 256) {
        float v = x2[row * NODES + i];
        s += v; s2 += v * v;
    }
    red[t] = s; red2[t] = s2;
    __syncthreads();
    for (int o = 128; o > 0; o >>= 1) {
        if (t < o) { red[t] += red[t + o]; red2[t] += red2[t + o]; }
        __syncthreads();
    }
    float mu  = red[0] * (1.f / NODES);
    float var = red2[0] * (1.f / NODES) - mu * mu;
    float rs  = rsqrtf(var + 1e-5f);
    for (int i = t; i < NODES; i += 256)
        out[row * NODES + i] = (x2[row * NODES + i] - mu) * rs * g[i] + b[i];
}

// ---------------------------------------------------------------------------
// tiny dense layers (16 rows): one wave per output element
// ---------------------------------------------------------------------------
__global__ __launch_bounds__(256)
void k_dense(const float* __restrict__ in, const float* __restrict__ W,
             const float* __restrict__ bias, float* __restrict__ out,
             int R, int K, int Ncol, int act)
{
    int wave = (blockIdx.x * blockDim.x + threadIdx.x) >> 5;
    int lane = threadIdx.x & 31;
    if (wave >= R * Ncol) return;
    int r = wave / Ncol, j = wave - r * Ncol;
    float s = 0.f;
    for (int k = lane; k < K; k += 32) s += in[r * K + k] * W[k * Ncol + j];
    for (int o = 16; o > 0; o >>= 1) s += __shfl_xor(s, o);
    if (lane == 0) {
        float v = s + bias[j];
        if (act) v = eluf(v);
        out[r * Ncol + j] = v;
    }
}

// ---------------------------------------------------------------------------
extern "C" void kernel_launch(void* const* d_in, const int* in_sizes, int n_in,
                              void* d_out, int out_size, void* d_ws, size_t ws_size,
                              hipStream_t stream)
{
    (void)in_sizes; (void)n_in; (void)out_size; (void)ws_size;

    const float* x   = (const float*)d_in[0];
    const int*   ei  = (const int*)d_in[1];
    /* d_in[2] = batch: layout sorted, unused */
    const float* W1  = (const float*)d_in[3];
    const float* as1 = (const float*)d_in[4];
    const float* ad1 = (const float*)d_in[5];
    const float* b1  = (const float*)d_in[6];
    const float* W2  = (const float*)d_in[7];
    const float* as2 = (const float*)d_in[8];
    const float* ad2 = (const float*)d_in[9];
    const float* b2  = (const float*)d_in[10];
    /* d_in[11]/[12] = p1w/p1b: dead (only layer3 used) */
    const float* p2w = (const float*)d_in[13];
    const float* p2b = (const float*)d_in[14];
    const float* lng = (const float*)d_in[15];
    const float* lnb = (const float*)d_in[16];
    const float* fw1 = (const float*)d_in[17];
    const float* fb1 = (const float*)d_in[18];
    const float* fw2 = (const float*)d_in[19];
    const float* fb2 = (const float*)d_in[20];
    const float* fw3 = (const float*)d_in[21];
    const float* fb3 = (const float*)d_in[22];
    const float* fw4 = (const float*)d_in[23];
    const float* fb4 = (const float*)d_in[24];
    const float* lw  = (const float*)d_in[25];
    const float* lb  = (const float*)d_in[26];

    char* wsp = (char*)d_ws;
    size_t off = 0;
    auto alloc = [&](size_t bytes) {
        char* p = wsp + off;
        off += (bytes + 255) & ~(size_t)255;
        return p;
    };
    float*  P    = (float*)alloc((size_t)NN * HCH * 4);   // GEMM out (both layers)
    float*  H1   = (float*)alloc((size_t)NN * HCH * 4);   // elu(gat1) f32
    float*  H2   = (float*)alloc((size_t)NN * HCH * 4);   // elu(gat2) f32
    bf16_t* Ab   = (bf16_t*)alloc((size_t)NN * DIN * 2);  // bf16(x)
    bf16_t* H1b  = (bf16_t*)alloc((size_t)NN * HCH * 2);  // bf16(h1)
    bf16_t* Wt1  = (bf16_t*)alloc((size_t)HCH * DIN * 2); // W1^T bf16 [512][1024]
    bf16_t* Wt2  = (bf16_t*)alloc((size_t)HCH * HCH * 2); // W2^T bf16 [512][512]
    float*  asr  = (float*)alloc((size_t)NN * NHEAD * 4);
    float*  adt  = (float*)alloc((size_t)NN * NHEAD * 4);
    int*   counts = (int*)alloc((size_t)NN * 4);
    int*   rowptr = (int*)alloc((size_t)(NN + 1) * 4);
    int*   cursor = (int*)alloc((size_t)NN * 4);
    int*   colidx = (int*)alloc((size_t)NEP * 4);
    float* x2 = (float*)alloc((size_t)NN * 4);
    float* e1 = (float*)alloc((size_t)NBATCH * 512 * 4);
    float* e2 = (float*)alloc((size_t)NBATCH * 256 * 4);
    float* e3 = (float*)alloc((size_t)NBATCH * 128 * 4);

    float* out  = (float*)d_out;              // [0,16384)      multiscale
    float* eout = out + NBATCH * NODES;       // [16384,17408)  e
    float* pred = eout + NBATCH * 64;         // [17408,17424)  pred

    // ---- CSR build (self loops via counts init = 1) ----
    k_init_counts  <<<NN / 256, 256, 0, stream>>>(counts);
    k_count_edges  <<<NE / 256, 256, 0, stream>>>(ei, counts);
    k_scan16k      <<<1, 1024, 0, stream>>>(counts, rowptr, cursor);
    k_scatter_self <<<NN / 256, 256, 0, stream>>>(cursor, colidx);
    k_scatter_edges<<<NE / 256, 256, 0, stream>>>(ei, cursor, colidx);

    // ---- bf16 operand prep (raw-copyable TDM tiles) ----
    k_f32_to_bf16   <<<(NN * DIN / 8) / 256, 256, 0, stream>>>(x, Ab, NN * DIN / 8);
    k_transpose_bf16<<<dim3(HCH / 32, DIN / 32), dim3(32, 8), 0, stream>>>(W1, Wt1, DIN, HCH);
    k_transpose_bf16<<<dim3(HCH / 32, HCH / 32), dim3(32, 8), 0, stream>>>(W2, Wt2, HCH, HCH);

    // ---- GAT layer 1 ----
    wmma_gemm_bf16<HCH><<<dim3(HCH / BN, NN / BM), 256, 0, stream>>>(Ab, Wt1, P, DIN);
    k_gat_alpha    <<<NN * 32 / 256, 256, 0, stream>>>(P, as1, ad1, asr, adt);
    k_gat_aggregate<<<NN * NHEAD * 32 / 256, 256, 0, stream>>>(P, asr, adt, rowptr, colidx, b1, H1, H1b);

    // ---- GAT layer 2 ----
    wmma_gemm_bf16<HCH><<<dim3(HCH / BN, NN / BM), 256, 0, stream>>>(H1b, Wt2, P, HCH);
    k_gat_alpha    <<<NN * 32 / 256, 256, 0, stream>>>(P, as2, ad2, asr, adt);
    k_gat_aggregate<<<NN * NHEAD * 32 / 256, 256, 0, stream>>>(P, asr, adt, rowptr, colidx, b2, H2, (bf16_t*)nullptr);

    // ---- pool2 -> x2 [16,1024], layernorm -> multiscale (d_out) ----
    k_pool      <<<NN * 32 / 256, 256, 0, stream>>>(H2, p2w, p2b, x2);
    k_layernorm <<<NBATCH, 256, 0, stream>>>(x2, lng, lnb, out);

    // ---- encoder MLP (16 rows, wave-per-output) ----
    k_dense<<<(NBATCH * 512 * 32 + 255) / 256, 256, 0, stream>>>(out, fw1, fb1, e1, NBATCH, 1024, 512, 1);
    k_dense<<<(NBATCH * 256 * 32 + 255) / 256, 256, 0, stream>>>(e1, fw2, fb2, e2, NBATCH, 512, 256, 1);
    k_dense<<<(NBATCH * 128 * 32 + 255) / 256, 256, 0, stream>>>(e2, fw3, fb3, e3, NBATCH, 256, 128, 1);
    k_dense<<<(NBATCH * 64 * 32 + 255) / 256, 256, 0, stream>>>(e3, fw4, fb4, eout, NBATCH, 128, 64, 1);
    k_dense<<<(NBATCH * 1 * 32 + 255) / 256, 256, 0, stream>>>(eout, lw, lb, pred, NBATCH, 64, 1, 0);
}
